// QLoRALinear_77584289235304
// MI455X (gfx1250) — compile-verified, matrix operand order checked by reading, not run on hardware
//
#include <hip/hip_runtime.h>
#include <hip/hip_bf16.h>
#include <stdint.h>

// ---------- problem constants (from reference) ----------
#define MQ 8192      // B*S = 4*2048 tokens
#define KQ 4096      // IN
#define NQ 4096      // OUT
#define GROUPQ 16    // fp4 block-scale group
#define RQ 16        // lora rank
#define SCALINGQ 1.0f

// ---------- gemm tiling ----------
#define BM 128
#define BN 128
#define BK 64                 // two WMMA K-steps per staged tile
#define NT (KQ / BK)          // 64 main-loop iterations

typedef __attribute__((ext_vector_type(16))) __bf16 v16bf;
typedef __attribute__((ext_vector_type(8)))  float  v8f;
typedef __attribute__((ext_vector_type(4)))  unsigned int u32x4;
typedef __attribute__((ext_vector_type(8)))  int    i32x8;
typedef __attribute__((ext_vector_type(4)))  int    i32x4;

// fp4 e2m1 decode: magnitudes {0,.5,1,1.5,2,3,4,6} via byte LUT of doubled values
__device__ __forceinline__ float fp4_val(uint32_t c) {
  uint32_t m   = c & 7u;
  uint32_t lut = (m < 4u) ? 0x03020100u : 0x0C080604u;   // {0,1,2,3} / {4,6,8,12}
  float v = 0.5f * (float)((lut >> ((m & 3u) * 8u)) & 0xFFu);
  return (c & 8u) ? -v : v;
}

__device__ __forceinline__ uint32_t pack_bf16x2(float a, float b) {
  union { __bf16 h[2]; uint32_t u; } t;
  t.h[0] = (__bf16)a; t.h[1] = (__bf16)b;
  return t.u;
}

// =====================================================================
// Kernel 1: dequantize packed NVFP4 weights -> bf16 buffer (done once;
// 32MB result lives in workspace and fits MI455X's 192MB L2, so the
// GEMM's 64x re-reads of W tiles are L2-resident).
// =====================================================================
__global__ __launch_bounds__(256) void k_dequant_w(
    const int* __restrict__ packed, const float* __restrict__ scales,
    const float* __restrict__ gscale, __bf16* __restrict__ wq) {
  int g   = blockIdx.x * 256 + threadIdx.x;   // group id, NQ*(KQ/16) total
  int row = g >> 8;                            // KQ/GROUPQ = 256 groups/row
  int gb  = g & 255;
  float s = scales[row * (KQ / GROUPQ) + gb] * gscale[0];
  const int* p = packed + (size_t)row * (KQ / 2) + gb * 8;
  union { __bf16 h[16]; uint4 q[2]; } o;
#pragma unroll
  for (int b = 0; b < 8; ++b) {
    uint32_t byte = (uint32_t)p[b];
    o.h[2 * b]     = (__bf16)(fp4_val(byte & 0xFu) * s);
    o.h[2 * b + 1] = (__bf16)(fp4_val((byte >> 4) & 0xFu) * s);
  }
  uint4* dst = (uint4*)(wq + (size_t)row * KQ + gb * 16);
  dst[0] = o.q[0];
  dst[1] = o.q[1];
}

// =====================================================================
// Kernel 2: T = SCALING * x @ lora_A^T  (M x 16), stored bf16 padded to
// 32 columns (cols 16..31 = 0) so the main GEMM consumes it as one
// extra K-chunk.
// =====================================================================
__global__ __launch_bounds__(256) void k_lora_xa(
    const float* __restrict__ x, const float* __restrict__ lA,
    __bf16* __restrict__ tq) {
  __shared__ float xs[16 * 64];
  __shared__ float as[16 * 64];
  const int tid = threadIdx.x;
  const int row = tid >> 4;          // 0..15 (token row within block)
  const int r   = tid & 15;          // 0..15 (lora rank)
  const int m0  = blockIdx.x * 16;
  const int lr  = tid >> 4;          // staging row
  const int lc  = (tid & 15) * 4;    // staging col (float4)
  float acc = 0.0f;
  for (int kc = 0; kc < KQ; kc += 64) {
    __syncthreads();
    *(float4*)&xs[lr * 64 + lc] = *(const float4*)(x + (size_t)(m0 + lr) * KQ + kc + lc);
    *(float4*)&as[lr * 64 + lc] = *(const float4*)(lA + (size_t)lr * KQ + kc + lc);
    __syncthreads();
#pragma unroll 8
    for (int kk = 0; kk < 64; ++kk) acc += xs[row * 64 + kk] * as[r * 64 + kk];
  }
  tq[(size_t)(m0 + row) * 32 + r]      = (__bf16)(acc * SCALINGQ);
  tq[(size_t)(m0 + row) * 32 + 16 + r] = (__bf16)0.0f;
}

// =====================================================================
// Kernel 3: main GEMM with TDM-prefetched, double-buffered LDS tiles.
// =====================================================================

// 16x32 bf16 WMMA operand fragment from a row-major LDS tile.
// ISA layout: lane<16 -> K{0..7,16..23}, lane>=16 -> K{8..15,24..31}.
struct FragBits { uint4 a, b; };
__device__ __forceinline__ v16bf frag_ld(const __bf16* tile, int row, int e0,
                                         int stride) {
  FragBits t;
  const __bf16* p = tile + row * stride + e0;
  t.a = *(const uint4*)p;            // K e0..e0+7
  t.b = *(const uint4*)(p + 16);     // K e0+16..e0+23
  return __builtin_bit_cast(v16bf, t);
}

// TDM 2D tile load (rows x cols bf16) -> LDS. D# per CDNA5 ISA 8.3/8.4.
__device__ __forceinline__ void tdm_load_2d(uint32_t lds_off, uint64_t gaddr,
                                            int cols, int rows,
                                            int tensor_d0, int tensor_d1,
                                            int row_stride) {
  u32x4 g0;
  g0[0] = 1u;                                        // count=1, user mode
  g0[1] = lds_off;                                   // lds_addr (bytes)
  g0[2] = (uint32_t)gaddr;                           // global_addr[31:0]
  g0[3] = (uint32_t)(gaddr >> 32) | (2u << 30);      // addr[56:32] | type=2
  i32x8 g1;
  g1[0] = (1 << 16);                                 // data_size=1 (2 bytes)
  g1[1] = (tensor_d0 & 0xFFFF) << 16;                // tensor_dim0 lo16
  g1[2] = ((tensor_d0 >> 16) & 0xFFFF) | ((tensor_d1 & 0xFFFF) << 16);
  g1[3] = ((tensor_d1 >> 16) & 0xFFFF) | (cols << 16);  // tile_dim0
  g1[4] = rows;                                      // tile_dim1 (tile_dim2=0)
  g1[5] = row_stride;                                // tensor_dim0_stride lo32
  g1[6] = 0;
  g1[7] = 0;
  i32x4 z4 = {0, 0, 0, 0};
  i32x8 z8 = {0, 0, 0, 0, 0, 0, 0, 0};
  __builtin_amdgcn_tensor_load_to_lds(g0, g1, z4, z4, z8, 0);
}

// 32 f32 -> 32 bf16 staging payload (one thread's share of an X tile)
struct XStage { uint4 q[4]; };
__device__ __forceinline__ XStage x_load_cvt(const float* xp) {
  XStage o;
#pragma unroll
  for (int q = 0; q < 4; ++q) {
    float4 a = *(const float4*)(xp + 8 * q);
    float4 b = *(const float4*)(xp + 8 * q + 4);
    o.q[q].x = pack_bf16x2(a.x, a.y);
    o.q[q].y = pack_bf16x2(a.z, a.w);
    o.q[q].z = pack_bf16x2(b.x, b.y);
    o.q[q].w = pack_bf16x2(b.z, b.w);
  }
  return o;
}
__device__ __forceinline__ void x_store(__bf16* dst, const XStage& s) {
#pragma unroll
  for (int q = 0; q < 4; ++q) *(uint4*)(dst + 8 * q) = s.q[q];
}

__global__ __launch_bounds__(256) void k_gemm(
    const float* __restrict__ x, const __bf16* __restrict__ wq,
    const __bf16* __restrict__ tq, const float* __restrict__ loraB,
    float* __restrict__ out) {
  __shared__ __bf16 Xs[2][BM * BK];   // 2 x 16 KB activations (ping-pong)
  __shared__ __bf16 Ws[2][BN * BK];   // 2 x 16 KB weights (TDM target)

  const int tid  = threadIdx.x;
  const int lane = tid & 31;
  const int wid  = tid >> 5;
  const int wm   = wid & 1;        // wave M index -> 64 rows
  const int wn   = wid >> 1;       // wave N index -> 32 cols
  const int m_block = blockIdx.y * BM;
  const int n_block = blockIdx.x * BN;

  const int lrow = lane & 15;
  const int e0   = (lane >> 4) * 8;      // fragment K element offset (0 or 8)

  const v8f zero8 = {0.f, 0.f, 0.f, 0.f, 0.f, 0.f, 0.f, 0.f};
  v8f acc[4][2];
#pragma unroll
  for (int i = 0; i < 4; ++i)
#pragma unroll
    for (int j = 0; j < 2; ++j) acc[i][j] = zero8;

  // X staging: thread covers half a 64-elem row (32 f32 -> 32 bf16)
  const int xrow = tid >> 1;
  const int xcol = (tid & 1) * 32;
  const float* xbase = x + (size_t)(m_block + xrow) * KQ + xcol;
  uint32_t lds_w[2], lds_x0;
  lds_w[0] = (uint32_t)(uintptr_t)&Ws[0][0];
  lds_w[1] = (uint32_t)(uintptr_t)&Ws[1][0];
  lds_x0   = (uint32_t)(uintptr_t)&Xs[0][0];
  const __bf16* wbase = wq + (size_t)n_block * KQ;

  // ---- prologue: stage tile 0 into buffer 0 ----
  if (tid == 0)
    tdm_load_2d(lds_w[0], (uint64_t)(uintptr_t)wbase, BK, BN, KQ, NQ, KQ);
  {
    XStage s0 = x_load_cvt(xbase);
    x_store(&Xs[0][xrow * BK + xcol], s0);
  }
  if (tid == 0) __builtin_amdgcn_s_wait_tensorcnt(0);
  __syncthreads();

  // ---- main loop: prefetch tile it+1 under compute of tile it ----
  for (int it = 0; it < NT; ++it) {
    const int cur = it & 1, nxt = cur ^ 1;
    const int k_next = (it + 1) * BK;
    const bool has_next = (it + 1) < NT;

    if (has_next && tid == 0)
      tdm_load_2d(lds_w[nxt], (uint64_t)(uintptr_t)(wbase + k_next),
                  BK, BN, KQ, NQ, KQ);
    XStage sx = {};
    if (has_next) sx = x_load_cvt(xbase + k_next);

    // two WMMA K-steps from the current buffers
#pragma unroll
    for (int s = 0; s < 2; ++s) {
      const int eo = s * 32 + e0;
      v16bf af[4], bfr[2];
#pragma unroll
      for (int i = 0; i < 4; ++i)
        af[i] = frag_ld(&Xs[cur][0], wm * 64 + i * 16 + lrow, eo, BK);
#pragma unroll
      for (int j = 0; j < 2; ++j)
        bfr[j] = frag_ld(&Ws[cur][0], wn * 32 + j * 16 + lrow, eo, BK);
#pragma unroll
      for (int i = 0; i < 4; ++i)
#pragma unroll
        for (int j = 0; j < 2; ++j)
          acc[i][j] = __builtin_amdgcn_wmma_f32_16x16x32_bf16(
              false, af[i], false, bfr[j], (short)0, acc[i][j], false, false);
    }

    if (has_next) x_store(&Xs[nxt][xrow * BK + xcol], sx);
    if (tid == 0) __builtin_amdgcn_s_wait_tensorcnt(0);  // prefetch landed
    __syncthreads();
  }

  // ---- LoRA epilogue: one 32-wide chunk [T | 0] x [lora_B | 0] ----
  if (tid == 0)   // TDM the M x 32 bf16 T tile (contiguous) into Xs[0]
    tdm_load_2d(lds_x0, (uint64_t)(uintptr_t)(tq + (size_t)m_block * 32),
                32, BM, 32, MQ, 32);
  if (tid < BN) {  // stage lora_B column tile: 16 bf16 + 16 zeros per row
    const float* bp = loraB + (size_t)(n_block + tid) * RQ;
    float4 b0 = *(const float4*)(bp + 0);
    float4 b1 = *(const float4*)(bp + 4);
    float4 b2 = *(const float4*)(bp + 8);
    float4 b3 = *(const float4*)(bp + 12);
    uint4 lo, hi;
    lo.x = pack_bf16x2(b0.x, b0.y); lo.y = pack_bf16x2(b0.z, b0.w);
    lo.z = pack_bf16x2(b1.x, b1.y); lo.w = pack_bf16x2(b1.z, b1.w);
    hi.x = pack_bf16x2(b2.x, b2.y); hi.y = pack_bf16x2(b2.z, b2.w);
    hi.z = pack_bf16x2(b3.x, b3.y); hi.w = pack_bf16x2(b3.z, b3.w);
    __bf16* dst = &Ws[0][tid * 32];
    *(uint4*)dst = lo;
    *(uint4*)(dst + 8) = hi;
    uint4 z = {0u, 0u, 0u, 0u};
    *(uint4*)(dst + 16) = z;
    *(uint4*)(dst + 24) = z;
  }
  if (tid == 0) __builtin_amdgcn_s_wait_tensorcnt(0);
  __syncthreads();
  {
    v16bf af[4], bfr[2];
#pragma unroll
    for (int i = 0; i < 4; ++i)
      af[i] = frag_ld(&Xs[0][0], wm * 64 + i * 16 + lrow, e0, 32);
#pragma unroll
    for (int j = 0; j < 2; ++j)
      bfr[j] = frag_ld(&Ws[0][0], wn * 32 + j * 16 + lrow, e0, 32);
#pragma unroll
    for (int i = 0; i < 4; ++i)
#pragma unroll
      for (int j = 0; j < 2; ++j)
        acc[i][j] = __builtin_amdgcn_wmma_f32_16x16x32_bf16(
            false, af[i], false, bfr[j], (short)0, acc[i][j], false, false);
  }

  // ---- store: C/D layout -> lane<16: M=r, lane>=16: M=r+8; N=lane&15 ----
#pragma unroll
  for (int i = 0; i < 4; ++i) {
    int mrow = m_block + wm * 64 + i * 16 + (lane >> 4) * 8;
#pragma unroll
    for (int j = 0; j < 2; ++j) {
      int ncol = n_block + wn * 32 + j * 16 + (lane & 15);
      float* op = out + (size_t)mrow * NQ + ncol;
#pragma unroll
      for (int r = 0; r < 8; ++r) op[(size_t)r * NQ] = acc[i][j][r];
    }
  }
}

// =====================================================================
extern "C" void kernel_launch(void* const* d_in, const int* in_sizes, int n_in,
                              void* d_out, int out_size, void* d_ws, size_t ws_size,
                              hipStream_t stream) {
  const float* x  = (const float*)d_in[0];
  const int*   pw = (const int*)d_in[1];
  const float* bs = (const float*)d_in[2];
  const float* gs = (const float*)d_in[3];
  const float* lA = (const float*)d_in[4];
  const float* lB = (const float*)d_in[5];
  float* out = (float*)d_out;

  // workspace: [0, 32MB) = bf16 dequantized W ; [32MB, +512KB) = bf16 T (Mx32)
  __bf16* wq = (__bf16*)d_ws;
  __bf16* tq = (__bf16*)((char*)d_ws + (size_t)NQ * KQ * 2);

  k_dequant_w<<<(NQ * (KQ / GROUPQ)) / 256, 256, 0, stream>>>(pw, bs, gs, wq);
  k_lora_xa<<<MQ / 16, 256, 0, stream>>>(x, lA, tq);
  dim3 grid(NQ / BN, MQ / BM);
  k_gemm<<<grid, 256, 0, stream>>>(x, wq, tq, lB, out);
}
